// TemporalGNNAnomalyDetector_17360257811028
// MI455X (gfx1250) — matrix operand order repaired and checked by reading, not compiled, not run on hardware
//
#include <hip/hip_runtime.h>

typedef __bf16 v16bf __attribute__((ext_vector_type(16)));
typedef __bf16 v8bf  __attribute__((ext_vector_type(8)));
typedef float  v8f   __attribute__((ext_vector_type(8)));

#define BN_EPS 1e-5f

// float -> bf16 (RNE) and bf16 -> float via bit ops (no fptrunc/fpext dependence)
static __device__ __forceinline__ __bf16 f2bf(float f) {
  unsigned u = __builtin_bit_cast(unsigned, f);
  u += 0x7FFFu + ((u >> 16) & 1u);
  unsigned short s = (unsigned short)(u >> 16);
  return __builtin_bit_cast(__bf16, s);
}
static __device__ __forceinline__ float bf2f(__bf16 b) {
  unsigned u = (unsigned)__builtin_bit_cast(unsigned short, b) << 16;
  return __builtin_bit_cast(float, u);
}

// two contiguous 16B runs of 8 bf16 -> one 16-element fragment (pure register naming)
static __device__ __forceinline__ v16bf load_frag(const __bf16* __restrict__ p0,
                                                  const __bf16* __restrict__ p1) {
  v8bf lo = *(const v8bf*)p0;
  v8bf hi = *(const v8bf*)p1;
  return __builtin_shufflevector(lo, hi, 0, 1, 2, 3, 4, 5, 6, 7,
                                 8, 9, 10, 11, 12, 13, 14, 15);
}

// D[N,OUT] = epilogue( A @ W^T (+ A2 @ W2^T) ); all matrix inputs bf16 row-major.
// A fragment (16x32): lane=half*16+mrow holds row M=mrow; elems 0..7 -> K=k0+half*8+e, 8..15 -> +16.
// B fragment (32x16): B[k][n]=W[n][k]; lane holds col N=mrow; elems 0..7 -> K=k0+half*8+e, 8..15 -> +16.
// One wave computes a 16 x (16*NT) strip. blockDim.x = 32*ceil(OUT/(16*NT)).
template <int NT>
__global__ void wmma_gemm_bf16_kernel(
    const __bf16* __restrict__ A, int lda,
    const __bf16* __restrict__ A2, int lda2,
    const __bf16* __restrict__ W, const __bf16* __restrict__ W2,  // [OUT,K] bf16
    const float* __restrict__ bias,
    const float* __restrict__ bng, const float* __restrict__ bnb,
    const float* __restrict__ bnm, const float* __restrict__ bnv,
    const __bf16* __restrict__ res, int ldres,                    // bf16 residual
    float* __restrict__ Df, int lddf,                             // optional fp32 out
    __bf16* __restrict__ Db, int lddb,                            // optional bf16 out
    int Nrows, int K, int K2, int OUT, int do_relu) {
  const int lane = threadIdx.x & 31;
  const int wave = threadIdx.x >> 5;
  const int half = lane >> 4;
  const int mrow = lane & 15;
  const int row0 = blockIdx.x * 16;
  const int colbase = wave * (16 * NT);
  int arow = row0 + mrow;
  if (arow >= Nrows) arow = Nrows - 1;

  v8f acc[NT];
#pragma unroll
  for (int t = 0; t < NT; ++t) acc[t] = (v8f){0.f, 0.f, 0.f, 0.f, 0.f, 0.f, 0.f, 0.f};

  const __bf16* ap = A + (long)arow * lda + half * 8;
  for (int k0 = 0; k0 < K; k0 += 32, ap += 32) {
    v16bf af = load_frag(ap, ap + 16);
#pragma unroll
    for (int t = 0; t < NT; ++t) {
      int col = colbase + t * 16 + mrow;
      int cc = col < OUT ? col : OUT - 1;  // padded cols never stored
      const __bf16* bp = W + (long)cc * K + k0 + half * 8;
      v16bf bf = load_frag(bp, bp + 16);
      acc[t] = __builtin_amdgcn_wmma_f32_16x16x32_bf16(false, af, false, bf,
                                                       (short)0, acc[t], false, false);
    }
  }
  if (A2 != nullptr) {  // uniform branch: EXEC unchanged around WMMA
    const __bf16* ap2 = A2 + (long)arow * lda2 + half * 8;
    for (int k0 = 0; k0 < K2; k0 += 32, ap2 += 32) {
      v16bf af = load_frag(ap2, ap2 + 16);
#pragma unroll
      for (int t = 0; t < NT; ++t) {
        int col = colbase + t * 16 + mrow;
        int cc = col < OUT ? col : OUT - 1;
        const __bf16* bp = W2 + (long)cc * K2 + k0 + half * 8;
        v16bf bf = load_frag(bp, bp + 16);
        acc[t] = __builtin_amdgcn_wmma_f32_16x16x32_bf16(false, af, false, bf,
                                                         (short)0, acc[t], false, false);
      }
    }
  }

  // C/D layout: VGPR v, lanes 0-15 -> M=v, lanes 16-31 -> M=v+8; N = lane&15
#pragma unroll
  for (int t = 0; t < NT; ++t) {
    int c = colbase + t * 16 + mrow;
#pragma unroll
    for (int v = 0; v < 8; ++v) {
      int r = row0 + v + half * 8;
      if (r < Nrows && c < OUT) {
        float val = acc[t][v];
        if (bias) val += bias[c];
        if (bng)  val = bng[c] * (val - bnm[c]) * rsqrtf(bnv[c] + BN_EPS) + bnb[c];
        if (do_relu) val = fmaxf(val, 0.0f);
        if (res) val += bf2f(res[(long)r * ldres + c]);
        if (Df) Df[(long)r * lddf + c] = val;
        if (Db) Db[(long)r * lddb + c] = f2bf(val);
      }
    }
  }
}

// fp32 [rows,srcK] -> bf16 [rows,dstK], zero-padded cols srcK..dstK-1
__global__ void cvt_pad_kernel(const float* __restrict__ src, int srcK,
                               __bf16* __restrict__ dst, int dstK, long total) {
  long i = (long)blockIdx.x * blockDim.x + threadIdx.x;
  if (i >= total) return;
  long r = i / dstK;
  int c = (int)(i - r * dstK);
  float v = (c < srcK) ? src[r * srcK + c] : 0.0f;
  dst[i] = f2bf(v);
}

__global__ void zero_f32_kernel(float* __restrict__ p, long n) {
  long i = (long)blockIdx.x * blockDim.x + threadIdx.x;
  if (i < n) p[i] = 0.0f;
}

// Edge gather (bf16, 16B per thread) + fp32 scatter-add; L2-resident feature map
__global__ void sage_scatter_kernel(const __bf16* __restrict__ h, int lda,
                                    const long long* __restrict__ src,
                                    const long long* __restrict__ dst,
                                    float* __restrict__ agg, float* __restrict__ deg,
                                    long E) {
  long t = (long)blockIdx.x * blockDim.x + threadIdx.x;
  if (t >= E * 8) return;
  long e = t >> 3;
  int g = (int)(t & 7) * 8;
  long s = src[e];
  long d = dst[e];
  v8bf v = *(const v8bf*)(h + s * (long)lda + g);
  float* ap = agg + d * 64 + g;
#pragma unroll
  for (int j = 0; j < 8; ++j) atomicAdd(ap + j, bf2f(v[j]));
  if (g == 0) atomicAdd(deg + d, 1.0f);
}

// SAGE mean: aggbf = bf16(agg / max(deg,1))
__global__ void agg_mean_bf_kernel(const float* __restrict__ agg, const float* __restrict__ deg,
                                   __bf16* __restrict__ out, long N) {
  long i = (long)blockIdx.x * blockDim.x + threadIdx.x;
  if (i >= N * 64) return;
  long n = i >> 6;
  out[i] = f2bf(agg[i] / fmaxf(deg[n], 1.0f));
}

// GRUCell gates (torch order r, z, n): h = (1-z)*n + z*h_prev ; dual fp32/bf16 output
__global__ void gru_gates_kernel(const float* __restrict__ gx, const float* __restrict__ gh,
                                 const float* __restrict__ h_prev, float* __restrict__ h_out,
                                 __bf16* __restrict__ hbf_out, long N) {
  long i = (long)blockIdx.x * blockDim.x + threadIdx.x;
  if (i >= N * 64) return;
  long n = i >> 6;
  int j = (int)(i & 63);
  const float* gxr = gx + n * 192;
  const float* ghr = gh + n * 192;
  float xr = gxr[j], xz = gxr[64 + j], xn = gxr[128 + j];
  float hr = ghr[j], hz = ghr[64 + j], hn = ghr[128 + j];
  float r = 1.0f / (1.0f + __expf(-(xr + hr)));
  float z = 1.0f / (1.0f + __expf(-(xz + hz)));
  float nn = tanhf(xn + r * hn);
  float hp = h_prev[n * 64 + j];
  float h = (1.0f - z) * nn + z * hp;
  h_out[i] = h;
  hbf_out[i] = f2bf(h);
}

extern "C" void kernel_launch(void* const* d_in, const int* in_sizes, int n_in,
                              void* d_out, int out_size, void* d_ws, size_t ws_size,
                              hipStream_t stream) {
  (void)n_in; (void)out_size; (void)ws_size;

  const float*     x      = (const float*)d_in[0];
  const long long* ei     = (const long long*)d_in[1];  // int64 [2,E]
  const float*     h_prev = (const float*)d_in[2];

  const long N  = in_sizes[2] / 64;
  const long E  = in_sizes[1] / 2;
  const int  IN = (int)(in_sizes[0] / N);

  auto P = [&](int i) { return (const float*)d_in[i]; };
  int pi = 3;
  const float* w_in = P(pi++); const float* b_in = P(pi++);
  const float* slw[2]; const float* slb[2]; const float* srw[2];
  for (int i = 0; i < 2; ++i) { slw[i] = P(pi++); slb[i] = P(pi++); srw[i] = P(pi++); }
  const float* bng[2]; const float* bnb[2]; const float* bnm[2]; const float* bnv[2];
  for (int i = 0; i < 2; ++i) { bng[i] = P(pi++); bnb[i] = P(pi++); bnm[i] = P(pi++); bnv[i] = P(pi++); }
  const float* jk_w = P(pi++); const float* jk_b = P(pi++);
  const float* w_ih = P(pi++); const float* w_hh = P(pi++);
  const float* b_ih = P(pi++); const float* b_hh = P(pi++);
  const float* c_w1 = P(pi++); const float* c_b1 = P(pi++);
  const float* cbg = P(pi++); const float* cbb = P(pi++);
  const float* cbm = P(pi++); const float* cbv = P(pi++);
  const float* c_w2 = P(pi++); const float* c_b2 = P(pi++);
  const float* c_w3 = P(pi++); const float* c_b3 = P(pi++);

  // ---- workspace ----
  // fp32 pool (time-shared): graph phase uses agg[N*64]+deg[N]; GRU phase uses gx/gh[N*192 each]
  float* fpool = (float*)d_ws;
  float* agg = fpool;
  float* deg = fpool + N * 64;
  float* gx  = fpool;
  float* gh  = fpool + N * 192;
  // bf16 region after fp32 pool (N*384 floats)
  __bf16* bp = (__bf16*)(fpool + N * 384);
  __bf16* xbf    = bp; bp += N * 192;   // x padded 165->192
  __bf16* hcatb  = bp; bp += N * 192;   // [h0|h1|h2] column-concat, bf16
  __bf16* aggbf  = bp; bp += N * 64;
  __bf16* hprevb = bp; bp += N * 64;
  __bf16* hjkb   = bp; bp += N * 64;
  __bf16* hcurb  = bp; bp += N * 64;
  __bf16* t1b    = bp; bp += N * 64;
  __bf16* t2b    = bp; bp += N * 32;
  // bf16 weights
  __bf16* w_inb = bp; bp += 64 * 192;   // padded 165->192
  __bf16* slwb[2], *srwb[2];
  for (int i = 0; i < 2; ++i) { slwb[i] = bp; bp += 64 * 64; srwb[i] = bp; bp += 64 * 64; }
  __bf16* jkwb = bp; bp += 64 * 192;
  __bf16* wihb = bp; bp += 192 * 64;
  __bf16* whhb = bp; bp += 192 * 64;
  __bf16* c1b = bp; bp += 64 * 64;
  __bf16* c2b = bp; bp += 32 * 64;
  __bf16* c3b = bp; bp += 2 * 32;

  float* hout   = (float*)d_out;  // [N,64]
  float* logits = hout + N * 64;  // [N,2]

  const int rt = (int)((N + 15) / 16);

#define CVT(src_, srcK_, dst_, dstK_, rows_)                                            \
  do {                                                                                  \
    long tot_ = (long)(rows_) * (dstK_);                                                \
    cvt_pad_kernel<<<(int)((tot_ + 255) / 256), 256, 0, stream>>>(src_, srcK_, dst_,    \
                                                                  dstK_, tot_);        \
  } while (0)

  // ---- one-shot bf16 conversions ----
  CVT(x, IN, xbf, 192, N);
  CVT(h_prev, 64, hprevb, 64, N);
  CVT(w_in, IN, w_inb, 192, 64);
  for (int i = 0; i < 2; ++i) { CVT(slw[i], 64, slwb[i], 64, 64); CVT(srw[i], 64, srwb[i], 64, 64); }
  CVT(jk_w, 192, jkwb, 192, 64);
  CVT(w_ih, 64, wihb, 64, 192);
  CVT(w_hh, 64, whhb, 64, 192);
  CVT(c_w1, 64, c1b, 64, 64);
  CVT(c_w2, 64, c2b, 64, 32);
  CVT(c_w3, 32, c3b, 32, 2);
#undef CVT

  const __bf16* nb = nullptr;
  float* nf = nullptr;

  // 1) input projection + ReLU -> hcatb[:,0:64]  (K padded to 192, aligned path)
  wmma_gemm_bf16_kernel<4><<<rt, 32, 0, stream>>>(
      xbf, 192, nb, 0, w_inb, nb, b_in, nf, nf, nf, nf, nb, 0,
      nf, 0, hcatb, 192, (int)N, 192, 0, 64, 1);

  // 2) two SAGE layers
  for (int i = 0; i < 2; ++i) {
    long zn = N * 65;  // agg + deg contiguous
    zero_f32_kernel<<<(int)((zn + 255) / 256), 256, 0, stream>>>(agg, zn);
    long st = E * 8;
    sage_scatter_kernel<<<(int)((st + 255) / 256), 256, 0, stream>>>(
        hcatb + i * 64, 192, ei, ei + E, agg, deg, E);
    long mt = N * 64;
    agg_mean_bf_kernel<<<(int)((mt + 255) / 256), 256, 0, stream>>>(agg, deg, aggbf, N);
    // h_{i+1} = relu(bn( mean@Wl^T + bl + h_i@Wr^T )) + h_i
    wmma_gemm_bf16_kernel<4><<<rt, 32, 0, stream>>>(
        aggbf, 64, hcatb + i * 64, 192, slwb[i], srwb[i], slb[i],
        bng[i], bnb[i], bnm[i], bnv[i], hcatb + i * 64, 192,
        nf, 0, hcatb + (i + 1) * 64, 192, (int)N, 64, 64, 64, 1);
  }

  // 3) jumping knowledge: relu([h0|h1|h2] @ jk_w^T + b) -> bf16 only
  wmma_gemm_bf16_kernel<4><<<rt, 32, 0, stream>>>(
      hcatb, 192, nb, 0, jkwb, nb, jk_b, nf, nf, nf, nf, nb, 0,
      nf, 0, hjkb, 64, (int)N, 192, 0, 64, 1);

  // 4) GRU pre-activations (OUT=192 -> 3 waves/block) -> fp32 for gate math
  wmma_gemm_bf16_kernel<4><<<rt, 96, 0, stream>>>(
      hjkb, 64, nb, 0, wihb, nb, b_ih, nf, nf, nf, nf, nb, 0,
      gx, 192, nullptr, 0, (int)N, 64, 0, 192, 0);
  wmma_gemm_bf16_kernel<4><<<rt, 96, 0, stream>>>(
      hprevb, 64, nb, 0, whhb, nb, b_hh, nf, nf, nf, nf, nb, 0,
      gh, 192, nullptr, 0, (int)N, 64, 0, 192, 0);

  // 5) GRU gates -> h_curr (fp32 output #1) + bf16 mirror
  long gt = N * 64;
  gru_gates_kernel<<<(int)((gt + 255) / 256), 256, 0, stream>>>(gx, gh, h_prev, hout, hcurb, N);

  // 6) classifier head
  wmma_gemm_bf16_kernel<4><<<rt, 32, 0, stream>>>(
      hcurb, 64, nb, 0, c1b, nb, c_b1, cbg, cbb, cbm, cbv, nb, 0,
      nf, 0, t1b, 64, (int)N, 64, 0, 64, 1);
  wmma_gemm_bf16_kernel<2><<<rt, 32, 0, stream>>>(
      t1b, 64, nb, 0, c2b, nb, c_b2, nf, nf, nf, nf, nb, 0,
      nf, 0, t2b, 32, (int)N, 64, 0, 32, 1);
  wmma_gemm_bf16_kernel<1><<<rt, 32, 0, stream>>>(
      t2b, 32, nb, 0, c3b, nb, c_b3, nf, nf, nf, nf, nb, 0,
      logits, 2, nullptr, 0, (int)N, 32, 0, 2, 0);
}
